// SelfAttention_90486370992871
// MI455X (gfx1250) — matrix operand order repaired and checked by reading, not compiled, not run on hardware
//
#include <hip/hip_runtime.h>

// ---------------------------------------------------------------------------
// Self-attention (B=4, HW=4096, C=512, R=64) for MI455X (gfx1250, wave32).
// All GEMMs via v_wmma_f32_16x16x32_bf16 (bf16 in, f32 accumulate).
// All WMMA operands are loaded with contiguous 16B accesses:
//   - A fragments from row-major [M][K]
//   - B fragments from transposed [N][K] (weights pre-transposed, V stored
//     transposed by the V-projection GEMM)
// Flash-attention softmax is parallelized across all 8 waves with
// shfl_xor butterflies (no extra barriers) and b128 LDS traffic.
// ---------------------------------------------------------------------------

typedef __attribute__((ext_vector_type(16))) __bf16 v16bf;
typedef __attribute__((ext_vector_type(8)))  float  v8f;

constexpr int BATCH = 4;
constexpr int HWN   = 4096;   // H*W
constexpr int CD    = 512;    // channels
constexpr int RD    = 64;     // reduced channels

// ----- helpers --------------------------------------------------------------

__device__ __forceinline__ unsigned short f2bf(float f) {
    __bf16 b = (__bf16)f;                      // native v_cvt f32->bf16
    unsigned short s;
    __builtin_memcpy(&s, &b, 2);
    return s;
}

__device__ __forceinline__ v8f wmma_bf16(v16bf a, v16bf b, v8f c) {
    return __builtin_amdgcn_wmma_f32_16x16x32_bf16(
        /*neg_a=*/false, a, /*neg_b=*/false, b,
        /*c_mod=*/(short)0, c, /*reuse_a=*/false, /*reuse_b=*/false);
}

union Frag128x2 { uint4 u[2]; v16bf v; };

// A fragment (16x32 bf16, M x K) from row-major [*, ld] at (row0, k0).
// ISA layout: lanes 0-15 -> M; per half-wave VGPR0..3 = K 0..7 (+8h),
// VGPR4..7 = K 16..23 (+8h). Two contiguous 16B loads per lane.
__device__ __forceinline__ v16bf load_a_frag(const unsigned short* A, int ld,
                                             int row0, int k0) {
    int lane = threadIdx.x & 31;
    int m = lane & 15, h = lane >> 4;
    const unsigned short* p = A + (size_t)(row0 + m) * ld + k0 + 8 * h;
    Frag128x2 f;
    f.u[0] = *reinterpret_cast<const uint4*>(p);        // K = k0+8h+0..7
    f.u[1] = *reinterpret_cast<const uint4*>(p + 16);   // K = k0+16+8h+0..7
    return f.v;
}

// B fragment (32x16 bf16, K x N) from a TRANSPOSED matrix Bt[N][K] (ld = K
// stride). Element (kk, n) = Bt[n0+n][k0+kk]; per half-wave h elements 0..7 =
// K k0+8h+0..7, elements 8..15 = K k0+16+8h+0..7 (matches A fragment).
__device__ __forceinline__ v16bf load_bt_frag(const unsigned short* Bt, int ld,
                                              int n0, int k0) {
    int lane = threadIdx.x & 31;
    int n = lane & 15, h = lane >> 4;
    const unsigned short* p = Bt + (size_t)(n0 + n) * ld + k0 + 8 * h;
    Frag128x2 f;
    f.u[0] = *reinterpret_cast<const uint4*>(p);
    f.u[1] = *reinterpret_cast<const uint4*>(p + 16);
    return f.v;
}

// C/D 16x16 f32 tile, row-major bf16 store (scalar b16 per element).
__device__ __forceinline__ void store_c_bf16(unsigned short* C, int ld,
                                             int row0, int col0, v8f c) {
    int lane = threadIdx.x & 31;
    int n = lane & 15, h = lane >> 4;
#pragma unroll
    for (int j = 0; j < 8; ++j)
        C[(size_t)(row0 + j + 8 * h) * ld + col0 + n] = f2bf(c[j]);
}

// C/D tile stored TRANSPOSED into Ct[N][M] (ldT = M stride): per lane the 8
// rows are consecutive -> pack to one 16B store.
__device__ __forceinline__ void store_c_bf16_T(unsigned short* Ct, int ldT,
                                               int row0, int col0, v8f c) {
    int lane = threadIdx.x & 31;
    int n = lane & 15, h = lane >> 4;
    union { unsigned short e[8]; uint4 u; } pk;
#pragma unroll
    for (int j = 0; j < 8; ++j) pk.e[j] = f2bf(c[j]);
    *reinterpret_cast<uint4*>(Ct + (size_t)(col0 + n) * ldT + row0 + 8 * h) = pk.u;
}

__device__ __forceinline__ void store_c_final(float* C, const float* R, int ld,
                                              int row0, int col0, float g, v8f c) {
    int lane = threadIdx.x & 31;
    int n = lane & 15, h = lane >> 4;
#pragma unroll
    for (int j = 0; j < 8; ++j) {
        size_t idx = (size_t)(row0 + j + 8 * h) * ld + col0 + n;
        C[idx] = g * c[j] + R[idx];
    }
}

// ----- kernel 0: f32 -> bf16 convert (plain and transposing) ---------------

__global__ void k_cvt(const float* __restrict__ s, unsigned short* __restrict__ d, int n) {
    int i = blockIdx.x * blockDim.x + threadIdx.x;
    if (i < n) d[i] = f2bf(s[i]);
}

// src: K x N (row-major) -> dst: N x K (row-major) bf16
__global__ void k_cvt_t(const float* __restrict__ s, unsigned short* __restrict__ d,
                        int K, int N) {
    int i = blockIdx.x * blockDim.x + threadIdx.x;
    if (i < K * N) {
        int kk = i / N, n = i - kk * N;
        d[(size_t)n * K + kk] = f2bf(s[i]);
    }
}

// ----- generic WMMA GEMM: C[M,N] = A[M,K] @ Bt[N,K]^T -----------------------
// 128 threads = 4 waves (2x2), each wave a 32x32 tile -> workgroup 64x64 tile.
// CMODE 0: bf16 row-major out; 1: bf16 transposed out; 2: f32 gamma*C + resid.

template <int CMODE>
__global__ __launch_bounds__(128) void k_gemm(
    const unsigned short* __restrict__ A, size_t strideA,
    const unsigned short* __restrict__ Bt,
    unsigned short* __restrict__ Cb, size_t strideC,
    float* __restrict__ Cf, const float* __restrict__ resid,
    const float* __restrict__ gamma_p, int M, int N, int K) {
    int bt = blockIdx.z;
    const unsigned short* Ab = A + (size_t)bt * strideA;
    int w = threadIdx.x >> 5;
    int row0 = blockIdx.y * 64 + (w >> 1) * 32;
    int col0 = blockIdx.x * 64 + (w & 1) * 32;

    v8f c00 = {0,0,0,0,0,0,0,0}, c01 = c00, c10 = c00, c11 = c00;
    for (int k0 = 0; k0 < K; k0 += 32) {
        v16bf a0 = load_a_frag(Ab, K, row0,      k0);
        v16bf a1 = load_a_frag(Ab, K, row0 + 16, k0);
        v16bf b0 = load_bt_frag(Bt, K, col0,      k0);
        v16bf b1 = load_bt_frag(Bt, K, col0 + 16, k0);
        c00 = wmma_bf16(a0, b0, c00);
        c01 = wmma_bf16(a0, b1, c01);
        c10 = wmma_bf16(a1, b0, c10);
        c11 = wmma_bf16(a1, b1, c11);
    }
    if (CMODE == 2) {
        float g = *gamma_p;
        float* out = Cf + (size_t)bt * strideC;
        const float* rs = resid + (size_t)bt * strideC;
        store_c_final(out, rs, N, row0,      col0,      g, c00);
        store_c_final(out, rs, N, row0,      col0 + 16, g, c01);
        store_c_final(out, rs, N, row0 + 16, col0,      g, c10);
        store_c_final(out, rs, N, row0 + 16, col0 + 16, g, c11);
    } else if (CMODE == 1) {
        unsigned short* out = Cb + (size_t)bt * strideC;
        store_c_bf16_T(out, M, row0,      col0,      c00);
        store_c_bf16_T(out, M, row0,      col0 + 16, c01);
        store_c_bf16_T(out, M, row0 + 16, col0,      c10);
        store_c_bf16_T(out, M, row0 + 16, col0 + 16, c11);
    } else {
        unsigned short* out = Cb + (size_t)bt * strideC;
        store_c_bf16(out, N, row0,      col0,      c00);
        store_c_bf16(out, N, row0,      col0 + 16, c01);
        store_c_bf16(out, N, row0 + 16, col0,      c10);
        store_c_bf16(out, N, row0 + 16, col0 + 16, c11);
    }
}

// ----- flash attention ------------------------------------------------------
// One workgroup (8 waves / 256 threads) owns 32 query rows of one batch.
// Per 64-key chunk:
//   1) each wave computes one 16x16 score tile (waves tiled 2x4)  -> LDS
//   2) ALL 256 threads run the online-softmax update: thread = (row, 8-key
//      segment); row max/sum via 3-step shfl_xor butterfly over the 8-lane
//      group; scores read as 2x b128, exp(P) written as 1x b128
//   3) every wave rescales its accumulators and runs P(32x64) @ V(64x64)
//      for its private 64-channel slice; V is read transposed ([C][HW])
//      straight from global with contiguous 16B fragment loads.

__global__ __launch_bounds__(256) void k_attn(
    const unsigned short* __restrict__ qg, const unsigned short* __restrict__ kg,
    const unsigned short* __restrict__ vtg, unsigned short* __restrict__ ag) {
    int bt = blockIdx.y;
    int rowBase = blockIdx.x * 32;
    const unsigned short* q  = qg  + (size_t)bt * HWN * RD;
    const unsigned short* k  = kg  + (size_t)bt * HWN * RD;
    const unsigned short* vt = vtg + (size_t)bt * CD * HWN;   // [C][HW]
    unsigned short* outa = ag + (size_t)bt * HWN * CD;

    __shared__ float          sS[32 * 64];        // score chunk (f32)
    __shared__ unsigned short sP[32 * 64];        // exp(scores) (bf16)
    __shared__ float          rowM[32], rowL[32], rowA[32];

    int w = threadIdx.x >> 5;
    int lane = threadIdx.x & 31;
    int mt = w >> 2, nt = w & 3;                  // this wave's score tile
    int ch0 = w * 64;                             // this wave's channel slice

    // Q fragments stay in VGPRs for the whole kernel (K = 64 -> 2 frags).
    v16bf qa0 = load_a_frag(q, RD, rowBase + mt * 16, 0);
    v16bf qa1 = load_a_frag(q, RD, rowBase + mt * 16, 32);

    v8f zero = {0,0,0,0,0,0,0,0};
    v8f u[2][4];
#pragma unroll
    for (int i = 0; i < 2; ++i)
#pragma unroll
        for (int j = 0; j < 4; ++j) u[i][j] = zero;

    if (threadIdx.x < 32) { rowM[threadIdx.x] = -1e30f; rowL[threadIdx.x] = 0.0f; }
    __syncthreads();

    int h = lane >> 4, n = lane & 15;
    int srow = threadIdx.x >> 3;                  // softmax: row 0..31
    int sseg = (threadIdx.x & 7) * 8;             // softmax: 8-key segment

    for (int c = 0; c < HWN / 64; ++c) {
        int key0 = c * 64;

        // ---- S tile = Q(16x64) * K^T(64x16) / sqrt(R) ----
        // kb row-major [HW][R] is exactly the transposed-B layout (N=key,K=feat).
        v16bf kb0 = load_bt_frag(k, RD, key0 + nt * 16, 0);
        v16bf kb1 = load_bt_frag(k, RD, key0 + nt * 16, 32);
        v8f s = zero;
        s = wmma_bf16(qa0, kb0, s);
        s = wmma_bf16(qa1, kb1, s);
#pragma unroll
        for (int j = 0; j < 8; ++j)
            sS[(mt * 16 + j + 8 * h) * 64 + nt * 16 + n] = s[j] * 0.125f;
        __syncthreads();

        // ---- online softmax (all 256 threads; 8 lanes cooperate per row) ----
        {
            float4 s0 = *reinterpret_cast<const float4*>(&sS[srow * 64 + sseg]);
            float4 s1 = *reinterpret_cast<const float4*>(&sS[srow * 64 + sseg + 4]);
            float e0 = s0.x, e1 = s0.y, e2 = s0.z, e3 = s0.w;
            float e4 = s1.x, e5 = s1.y, e6 = s1.z, e7 = s1.w;
            float pmax = fmaxf(fmaxf(fmaxf(e0, e1), fmaxf(e2, e3)),
                               fmaxf(fmaxf(e4, e5), fmaxf(e6, e7)));
#pragma unroll
            for (int mk = 1; mk < 8; mk <<= 1)
                pmax = fmaxf(pmax, __shfl_xor(pmax, mk, 32));
            float m0 = rowM[srow];
            float mx = fmaxf(m0, pmax);
            float alpha = __expf(m0 - mx);
            e0 = __expf(e0 - mx); e1 = __expf(e1 - mx);
            e2 = __expf(e2 - mx); e3 = __expf(e3 - mx);
            e4 = __expf(e4 - mx); e5 = __expf(e5 - mx);
            e6 = __expf(e6 - mx); e7 = __expf(e7 - mx);
            float psum = ((e0 + e1) + (e2 + e3)) + ((e4 + e5) + (e6 + e7));
#pragma unroll
            for (int mk = 1; mk < 8; mk <<= 1)
                psum += __shfl_xor(psum, mk, 32);
            union { unsigned short e[8]; uint4 u; } pk;
            pk.e[0] = f2bf(e0); pk.e[1] = f2bf(e1);
            pk.e[2] = f2bf(e2); pk.e[3] = f2bf(e3);
            pk.e[4] = f2bf(e4); pk.e[5] = f2bf(e5);
            pk.e[6] = f2bf(e6); pk.e[7] = f2bf(e7);
            *reinterpret_cast<uint4*>(&sP[srow * 64 + sseg]) = pk.u;
            if ((threadIdx.x & 7) == 0) {
                rowL[srow] = rowL[srow] * alpha + psum;
                rowM[srow] = mx;
                rowA[srow] = alpha;
            }
        }
        __syncthreads();

        // ---- rescale running accumulators by alpha(row) ----
        float al[2][8];
#pragma unroll
        for (int m2 = 0; m2 < 2; ++m2)
#pragma unroll
            for (int j = 0; j < 8; ++j) al[m2][j] = rowA[m2 * 16 + j + 8 * h];
#pragma unroll
        for (int m2 = 0; m2 < 2; ++m2)
#pragma unroll
            for (int t = 0; t < 4; ++t)
#pragma unroll
                for (int j = 0; j < 8; ++j) u[m2][t][j] *= al[m2][j];

        // ---- U += P(32x64) @ V(64x64 slice), two K-steps of 32 ----
#pragma unroll
        for (int ks = 0; ks < 2; ++ks) {
            v16bf pa0 = load_a_frag(sP, 64, 0,  ks * 32);
            v16bf pa1 = load_a_frag(sP, 64, 16, ks * 32);
#pragma unroll
            for (int t = 0; t < 4; ++t) {
                // B element (kk = key offset, n = channel) = vt[ch][key]
                v16bf bf = load_bt_frag(vt, HWN, ch0 + t * 16, key0 + ks * 32);
                u[0][t] = wmma_bf16(pa0, bf, u[0][t]);
                u[1][t] = wmma_bf16(pa1, bf, u[1][t]);
            }
        }
        __syncthreads();   // protect sS / sP before next chunk
    }

    // ---- finalize: divide by row sums, write attended (bf16, row-major) ----
    float li[2][8];
#pragma unroll
    for (int m2 = 0; m2 < 2; ++m2)
#pragma unroll
        for (int j = 0; j < 8; ++j) li[m2][j] = 1.0f / rowL[m2 * 16 + j + 8 * h];
#pragma unroll
    for (int m2 = 0; m2 < 2; ++m2)
#pragma unroll
        for (int t = 0; t < 4; ++t)
#pragma unroll
            for (int j = 0; j < 8; ++j)
                outa[(size_t)(rowBase + m2 * 16 + j + 8 * h) * CD + ch0 + t * 16 + n] =
                    f2bf(u[m2][t][j] * li[m2][j]);
}

// ----- launch ---------------------------------------------------------------

extern "C" void kernel_launch(void* const* d_in, const int* in_sizes, int n_in,
                              void* d_out, int out_size, void* d_ws, size_t ws_size,
                              hipStream_t stream) {
    const float* x     = (const float*)d_in[0];
    const float* Wq    = (const float*)d_in[1];
    const float* Wk    = (const float*)d_in[2];
    const float* Wv    = (const float*)d_in[3];
    const float* Wo    = (const float*)d_in[4];
    const float* gamma = (const float*)d_in[5];

    char* p = (char*)d_ws;
    unsigned short* xb   = (unsigned short*)p;  p += (size_t)BATCH * HWN * CD * 2;
    unsigned short* Wqt  = (unsigned short*)p;  p += (size_t)CD * RD * 2;   // [R][C]
    unsigned short* Wkt  = (unsigned short*)p;  p += (size_t)CD * RD * 2;   // [R][C]
    unsigned short* Wvt  = (unsigned short*)p;  p += (size_t)CD * CD * 2;   // [C][C]
    unsigned short* Wot  = (unsigned short*)p;  p += (size_t)CD * CD * 2;   // [C][C]
    unsigned short* qb   = (unsigned short*)p;  p += (size_t)BATCH * HWN * RD * 2;
    unsigned short* kb   = (unsigned short*)p;  p += (size_t)BATCH * HWN * RD * 2;
    unsigned short* vbt  = (unsigned short*)p;  p += (size_t)BATCH * CD * HWN * 2; // [C][HW]
    unsigned short* ab   = (unsigned short*)p;  p += (size_t)BATCH * HWN * CD * 2;

    k_cvt<<<(BATCH * HWN * CD + 255) / 256, 256, 0, stream>>>(x, xb, BATCH * HWN * CD);
    k_cvt_t<<<(CD * RD + 255) / 256, 256, 0, stream>>>(Wq, Wqt, CD, RD);
    k_cvt_t<<<(CD * RD + 255) / 256, 256, 0, stream>>>(Wk, Wkt, CD, RD);
    k_cvt_t<<<(CD * CD + 255) / 256, 256, 0, stream>>>(Wv, Wvt, CD, CD);
    k_cvt_t<<<(CD * CD + 255) / 256, 256, 0, stream>>>(Wo, Wot, CD, CD);

    // q = xb @ Wq, k = xb @ Wk   (M=4096, N=64, K=512), row-major bf16 out
    k_gemm<0><<<dim3(1, HWN / 64, BATCH), 128, 0, stream>>>(
        xb, (size_t)HWN * CD, Wqt, qb, (size_t)HWN * RD,
        nullptr, nullptr, nullptr, HWN, RD, CD);
    k_gemm<0><<<dim3(1, HWN / 64, BATCH), 128, 0, stream>>>(
        xb, (size_t)HWN * CD, Wkt, kb, (size_t)HWN * RD,
        nullptr, nullptr, nullptr, HWN, RD, CD);
    // v = xb @ Wv  -> stored TRANSPOSED [C][HW] with packed b128 stores
    k_gemm<1><<<dim3(CD / 64, HWN / 64, BATCH), 128, 0, stream>>>(
        xb, (size_t)HWN * CD, Wvt, vbt, (size_t)CD * HWN,
        nullptr, nullptr, nullptr, HWN, CD, CD);

    // attended = softmax(q k^T / 8) @ v
    k_attn<<<dim3(HWN / 32, BATCH), 256, 0, stream>>>(qb, kb, vbt, ab);

    // out = gamma * (attended @ Wo) + x   (f32)
    k_gemm<2><<<dim3(CD / 64, HWN / 64, BATCH), 128, 0, stream>>>(
        ab, (size_t)HWN * CD, Wot, nullptr, (size_t)HWN * CD,
        (float*)d_out, x, gamma, HWN, CD, CD);
}
